// LinearFWLayer_38843684225809
// MI455X (gfx1250) — matrix operand order repaired and checked by reading, not compiled, not run on hardware
//
#include <hip/hip_runtime.h>
#include <hip/hip_bf16.h>
#include <cstdint>

typedef __attribute__((ext_vector_type(16))) __bf16 v16bf;
typedef __attribute__((ext_vector_type(8)))  float  v8f;

__device__ __forceinline__ unsigned short f2bf_bits(float f) {
  unsigned u = __float_as_uint(f);
  u += 0x7FFFu + ((u >> 16) & 1u);          // round-to-nearest-even
  return (unsigned short)(u >> 16);
}
__device__ __forceinline__ float bf2f(unsigned short b) {
  return __uint_as_float(((unsigned)b) << 16);
}

#define KCHUNK  128     // K elements staged per LDS buffer
#define BSTRIDE 136     // LDS row stride (bf16 elems): 272B, 16B-aligned, bank-conflict-free
#define NUMU    ((KCHUNK / 32) * 4)   // 16 WMMA-pair steps per chunk

// ---------------------------------------------------------------------------
// bf16 WMMA GEMM:  C[b] = A[b] (MxK, row-major) * Bt[b]^T  (Bt is NxK row-major)
// B strip (64 cols x KCHUNK) staged in LDS via async global->LDS copies,
// double-buffered: fill of chunk k+1 overlaps the 32 WMMAs of chunk k.
// B-fragments are software-pipelined in registers (ds_load for frag u+1 is
// issued before the WMMAs consuming frag u), so dscnt waits are covered.
// Wave tile: 32(M) x 64(N); 8 waves -> block tile 256 x 64.
// Epilogue: bias (per-N or per-M), phi (elu+1), per-row z scale, fp32 residual.
// ---------------------------------------------------------------------------
__global__ __launch_bounds__(256) void wmma_gemm_bf16(
    const unsigned short* __restrict__ A,  long long strA,
    const unsigned short* __restrict__ Bt, long long strB,
    int M, int N, int K,
    const float* __restrict__ bias, int bias_mode,   // 0=none 1=bias[n] 2=bias[m]
    int apply_phi,
    const float* __restrict__ zvec, long long strZ,
    const float* __restrict__ resid,
    unsigned short* __restrict__ Cbf, float* __restrict__ Cf,
    long long strC, int store_f32)
{
  __shared__ unsigned short ldsB[2][64 * BSTRIDE];

  const int b = blockIdx.z;
  A  += (long long)b * strA;
  Bt += (long long)b * strB;

  const int wave = threadIdx.x >> 5;
  const int lane = threadIdx.x & 31;
  const int row0 = blockIdx.y * 256 + wave * 32;
  const int col0 = blockIdx.x * 64;

  v8f acc[2][4] = {};

  // A-frag bases (ISA 16-bit A layout): lane<16 row=lane K{0..7,16..23};
  // lane>=16 same rows K{8..15,24..31}
  const unsigned short* ap0 = A + (long long)(row0 + (lane & 15)) * K + ((lane >> 4) << 3);
  const unsigned short* ap1 = ap0 + 16LL * K;

  // Async fill of one LDS buffer with Bt[col0..col0+63][k0..k0+KCHUNK):
  // 1024 x b128 transfers, 4 per thread, LDS-direct (ASYNCcnt-tracked).
  auto fill = [&](int buf, int k0) {
#pragma unroll
    for (int j = 0; j < 4; ++j) {
      const int lin = (int)threadIdx.x + 256 * j;  // 0..1023
      const int col = lin >> 4;                    // 0..63
      const int kk  = (lin & 15) << 3;             // 0,8,...,120
      const unsigned short* g = Bt + (long long)(col0 + col) * K + k0 + kk;
      const unsigned lds_off = (unsigned)(uintptr_t)&ldsB[buf][col * BSTRIDE + kk];
      asm volatile("global_load_async_to_lds_b128 %0, %1, off"
                   :: "v"(lds_off), "v"((unsigned long long)(uintptr_t)g)
                   : "memory");
    }
  };

  fill(0, 0);
  asm volatile("s_wait_asynccnt 0" ::: "memory");
  __syncthreads();

  int cur = 0;
  for (int k0 = 0; k0 < K; k0 += KCHUNK) {
    if (k0 + KCHUNK < K) fill(cur ^ 1, k0 + KCHUNK);   // overlap fill with compute

    __builtin_prefetch(ap0 + k0 + KCHUNK, 0, 1);
    __builtin_prefetch(ap1 + k0 + KCHUNK, 0, 1);

    // B-frag loader from current LDS buffer (16-bit B 32x16 layout matching A:
    // lane<16 col=lane K{0..7,16..23}; lane>=16 K{8..15,24..31})
    auto ldb = [&](int u) -> v16bf {
      const int ks = (u >> 2) * 32;
      const int t  = u & 3;
      const int c  = (lane & 15) + t * 16;
      const unsigned short* lp = &ldsB[cur][c * BSTRIDE + ks + ((lane >> 4) << 3)];
      v16bf f;
      uint4* fp = (uint4*)&f;
      fp[0] = *(const uint4*)(lp);
      fp[1] = *(const uint4*)(lp + 16);
      return f;
    };
    auto lda = [&](int ks, v16bf& a0, v16bf& a1) {
      uint4* fp = (uint4*)&a0;
      fp[0] = *(const uint4*)(ap0 + k0 + ks);
      fp[1] = *(const uint4*)(ap0 + k0 + ks + 16);
      uint4* gp = (uint4*)&a1;
      gp[0] = *(const uint4*)(ap1 + k0 + ks);
      gp[1] = *(const uint4*)(ap1 + k0 + ks + 16);
    };

    v16bf a0, a1, b_cur, b_nxt;
    lda(0, a0, a1);
    b_cur = ldb(0);

#pragma unroll
    for (int u = 0; u < NUMU; ++u) {
      const int t = u & 3;
      if (u + 1 < NUMU) b_nxt = ldb(u + 1);          // issue LDS load early
      acc[0][t] = __builtin_amdgcn_wmma_f32_16x16x32_bf16(
          false, a0, false, b_cur, (short)0, acc[0][t], false, false);
      acc[1][t] = __builtin_amdgcn_wmma_f32_16x16x32_bf16(
          false, a1, false, b_cur, (short)0, acc[1][t], false, false);
      b_cur = b_nxt;
      if (t == 3 && u + 1 < NUMU) lda(((u + 1) >> 2) * 32, a0, a1);  // next K-step A
    }

    asm volatile("s_wait_asynccnt 0" ::: "memory");
    __syncthreads();
    cur ^= 1;
  }

  // C layout: vgpr g, lane -> m = base + 8*(lane>=16) + g, n = col + (lane&15)
  const int nb = col0 + (lane & 15);
#pragma unroll
  for (int mt = 0; mt < 2; ++mt) {
    const int mbase = row0 + mt * 16 + ((lane >> 4) << 3);
#pragma unroll
    for (int t = 0; t < 4; ++t) {
      const int n = nb + t * 16;
      const float bn = (bias_mode == 1) ? bias[n] : 0.0f;
#pragma unroll
      for (int g = 0; g < 8; ++g) {
        const int m = mbase + g;
        float x = acc[mt][t][g] + ((bias_mode == 2) ? bias[m] : bn);
        if (apply_phi) x = (x > 0.0f) ? (x + 1.0f) : __expf(x);
        if (zvec) x *= zvec[(long long)b * strZ + m];
        const long long ci = (long long)b * strC + (long long)m * N + n;
        if (store_f32) {
          float r = resid ? resid[ci] : 0.0f;
          __builtin_nontemporal_store(x + r, &Cf[ci]);
        } else {
          __builtin_nontemporal_store(f2bf_bits(x), &Cbf[ci]);
        }
      }
    }
  }
}

// ---------------------------------------------------------------------------
// Elementwise fp32 -> bf16 convert
// ---------------------------------------------------------------------------
__global__ void cvt_f32_to_bf16(const float* __restrict__ in,
                                unsigned short* __restrict__ out, long long n) {
  long long i = (long long)blockIdx.x * blockDim.x + threadIdx.x;
  if (i < n) out[i] = f2bf_bits(in[i]);
}

// fp32 [R,C] -> bf16 [C,R] (weights: Bt layout)
__global__ void transpose_f32_to_bf16(const float* __restrict__ in,
                                      unsigned short* __restrict__ out, int R, int C) {
  __shared__ float tile[16][17];
  int r = blockIdx.y * 16 + threadIdx.y;
  int c = blockIdx.x * 16 + threadIdx.x;
  tile[threadIdx.y][threadIdx.x] = in[(long long)r * C + c];
  __syncthreads();
  int orow = blockIdx.x * 16 + threadIdx.y;
  int ocol = blockIdx.y * 16 + threadIdx.x;
  out[(long long)orow * R + ocol] = f2bf_bits(tile[threadIdx.x][threadIdx.y]);
}

// Row sums of bf16 matrix (one block per row): k_sum[b,d] = sum_s kT[b,d,s]
__global__ void rowsum_bf16(const unsigned short* __restrict__ in,
                            float* __restrict__ out, int Klen) {
  long long r = blockIdx.x;
  const unsigned short* p = in + r * (long long)Klen;
  float s = 0.0f;
  for (int i = threadIdx.x; i < Klen; i += 256) s += bf2f(p[i]);
  __shared__ float red[256];
  red[threadIdx.x] = s;
  __syncthreads();
  for (int off = 128; off > 0; off >>= 1) {
    if ((int)threadIdx.x < off) red[threadIdx.x] += red[threadIdx.x + off];
    __syncthreads();
  }
  if (threadIdx.x == 0) out[r] = red[0];
}

// z[b,s] = 1 / (sum_d qphi[b,s,d] * ksum[b,d] + 1e-6); one wave per row
__global__ void compute_z(const unsigned short* __restrict__ qphi,
                          const float* __restrict__ ksum,
                          float* __restrict__ z, int S, int D) {
  int row  = blockIdx.x * (blockDim.x >> 5) + (threadIdx.x >> 5);
  int lane = threadIdx.x & 31;
  int b    = row / S;
  const unsigned short* q = qphi + (long long)row * D;
  const float* ks = ksum + (long long)b * D;
  float s = 0.0f;
  for (int i = lane; i < D; i += 32) s += bf2f(q[i]) * ks[i];
  for (int off = 16; off > 0; off >>= 1) s += __shfl_xor(s, off, 32);
  if (lane == 0) z[row] = 1.0f / (s + 1e-6f);
}

// ---------------------------------------------------------------------------
extern "C" void kernel_launch(void* const* d_in, const int* in_sizes, int n_in,
                              void* d_out, int out_size, void* d_ws, size_t ws_size,
                              hipStream_t stream) {
  const int B = 4, S = 4096, D = 1024;
  const long long MS = (long long)B * S * D;   // 16,777,216 elements

  const float* inputs  = (const float*)d_in[0];
  const float* context = (const float*)d_in[1];
  const float* Wq = (const float*)d_in[2];
  const float* bq = (const float*)d_in[3];
  const float* Wk = (const float*)d_in[4];
  const float* bk = (const float*)d_in[5];
  const float* Wv = (const float*)d_in[6];
  const float* bv = (const float*)d_in[7];
  const float* Wo = (const float*)d_in[8];
  const float* bo = (const float*)d_in[9];
  float* out = (float*)d_out;

  char* ws = (char*)d_ws;
  const size_t MB = 1ull << 20;
  unsigned short* xin  = (unsigned short*)(ws + 0);        // 32 MB  bf16(inputs)  [B*S,D]
  unsigned short* xctx = (unsigned short*)(ws + 32*MB);    // 32 MB  bf16(context) [B*S,D]
  unsigned short* WqT  = (unsigned short*)(ws + 64*MB);    //  2 MB  Wq^T [N,K]
  unsigned short* WkT  = (unsigned short*)(ws + 66*MB);
  unsigned short* WvT  = (unsigned short*)(ws + 68*MB);
  unsigned short* WoT  = (unsigned short*)(ws + 70*MB);
  unsigned short* qphi = (unsigned short*)(ws + 72*MB);    // 32 MB  [B*S,D]
  unsigned short* kT   = (unsigned short*)(ws + 104*MB);   // 32 MB  [B,D,S] (phi(k) transposed)
  unsigned short* vT   = (unsigned short*)(ws + 136*MB);   // 32 MB  [B,D,S] (v transposed)
  unsigned short* kvT  = (unsigned short*)(ws + 168*MB);   //  8 MB  [B,D(e),D(d)]
  float*          ksum = (float*)(ws + 176*MB);            // 16 KB  [B,D]
  float*          zbuf = (float*)(ws + 177*MB);            // 64 KB  [B,S]
  unsigned short* oatt = (unsigned short*)(ws + 178*MB);   // 32 MB  [B*S,D]

  dim3 blk256(256);
  dim3 tb(16, 16);

  // 1) convert activations to bf16
  cvt_f32_to_bf16<<<(unsigned)(MS / 256), blk256, 0, stream>>>(inputs,  xin,  MS);
  cvt_f32_to_bf16<<<(unsigned)(MS / 256), blk256, 0, stream>>>(context, xctx, MS);

  // 2) transpose+convert weights -> [N,K] layout (used as Bt or as transposed-A)
  dim3 tgW(D / 16, D / 16, 1);
  transpose_f32_to_bf16<<<tgW, tb, 0, stream>>>(Wq, WqT, D, D);
  transpose_f32_to_bf16<<<tgW, tb, 0, stream>>>(Wk, WkT, D, D);
  transpose_f32_to_bf16<<<tgW, tb, 0, stream>>>(Wv, WvT, D, D);
  transpose_f32_to_bf16<<<tgW, tb, 0, stream>>>(Wo, WoT, D, D);

  // 3) q_phi[s,e] = phi(X Wq + bq)        (A = xin, Bt = WqT)
  dim3 gProj(D / 64, (B * S) / 256, 1);
  wmma_gemm_bf16<<<gProj, blk256, 0, stream>>>(xin, 0, WqT, 0, B * S, D, D,
      bq, 1, 1, nullptr, 0, nullptr, qphi, nullptr, 0, 0);

  // 4) kT[b][d][s] = phi(sum_e Wk[e,d] ctx[s,e] + bk[d])   (A = WkT, Bt = ctx[b])
  //    transposed projection directly, no separate transpose pass; bias per-M.
  dim3 gTP(S / 64, D / 256, B);
  wmma_gemm_bf16<<<gTP, blk256, 0, stream>>>(WkT, 0, xctx, (long long)S * D, D, S, D,
      bk, 2, 1, nullptr, 0, nullptr, kT, nullptr, (long long)D * S, 0);

  // 5) vT[b][e][s] = sum_d Wv[d,e] ctx[s,d] + bv[e]        (A = WvT, Bt = ctx[b])
  wmma_gemm_bf16<<<gTP, blk256, 0, stream>>>(WvT, 0, xctx, (long long)S * D, D, S, D,
      bv, 2, 0, nullptr, 0, nullptr, vT, nullptr, (long long)D * S, 0);

  // 6) kvT[b][e][d] = sum_s v[s,e] * k_phi[s,d]   (A = vT, Bt = kT, K = S)
  dim3 gKV(D / 64, D / 256, B);
  wmma_gemm_bf16<<<gKV, blk256, 0, stream>>>(vT, (long long)D * S, kT, (long long)D * S,
      D, D, S, nullptr, 0, 0, nullptr, 0, nullptr, kvT, nullptr, (long long)D * D, 0);

  // 7) k_sum[b,d] = sum_s k_phi[b,s,d] (row sums of kT)
  rowsum_bf16<<<B * D, blk256, 0, stream>>>(kT, ksum, S);

  // 8) z[b,s] = 1 / (q_phi . k_sum + 1e-6)
  compute_z<<<(B * S) / 8, blk256, 0, stream>>>(qphi, ksum, zbuf, S, D);

  // 9) oatt[b,s,e] = z[b,s] * sum_d qphi[s,d] kvT[e,d]
  dim3 gAtt(D / 64, S / 256, B);
  wmma_gemm_bf16<<<gAtt, blk256, 0, stream>>>(qphi, (long long)S * D, kvT, (long long)D * D,
      S, D, D, nullptr, 0, 0, zbuf, S, nullptr, oatt, nullptr, (long long)S * D, 0);

  // 10) out = oatt @ Wo + bo + inputs   (fp32 output)
  wmma_gemm_bf16<<<gProj, blk256, 0, stream>>>(oatt, 0, WoT, 0, B * S, D, D,
      bo, 1, 0, nullptr, 0, inputs, nullptr, out, 0, 1);
}
